// HCEN_83210696393590
// MI455X (gfx1250) — compile-verified, matrix operand order checked by reading, not run on hardware
//
#include <hip/hip_runtime.h>

// ---------------------------------------------------------------------------
// HCEN forward on MI455X (gfx1250):
//   m   = mean(x, axis=S)          -> bandwidth bound: 256MB / 23.3TB/s ~ 11us
//   enc = m @ W_enc^T + b_enc      -> tiny GEMM, WMMA f32 16x16x4
//   out = enc @ W_out^T + b_out    -> tiny GEMM, WMMA f32 16x16x4
// ---------------------------------------------------------------------------

#define BATCH  16
#define SEQ    4096
#define DIM    1024
#define SSPLIT 64                      // S-axis split for the reduction
#define SCHUNK (SEQ / SSPLIT)          // 64 rows per block

typedef float  v2f  __attribute__((ext_vector_type(2)));
typedef float  v4f  __attribute__((ext_vector_type(4)));
typedef float  v8f  __attribute__((ext_vector_type(8)));
typedef _Float16 v16h __attribute__((ext_vector_type(16)));

// ---------------------------------------------------------------------------
// Stage 1: each block = (s-chunk, batch). 256 threads x float4 columns.
// Streaming (non-temporal) b128 loads; x is read exactly once.
// Writes partial sums [SSPLIT][B][DIM] to workspace. Deterministic.
// ---------------------------------------------------------------------------
__global__ __launch_bounds__(256) void hcen_mean_stage1(
    const float* __restrict__ x, float* __restrict__ part) {
  const int sblk = blockIdx.x;          // 0..SSPLIT-1
  const int b    = blockIdx.y;          // 0..BATCH-1
  const int t    = threadIdx.x;         // 0..255, owns columns 4t..4t+3

  const float* xp = x + ((size_t)b * SEQ + (size_t)sblk * SCHUNK) * DIM + t * 4;
  v4f acc = {0.f, 0.f, 0.f, 0.f};
#pragma unroll 4
  for (int s = 0; s < SCHUNK; ++s) {
    v4f v = __builtin_nontemporal_load((const v4f*)(xp + (size_t)s * DIM));
    acc += v;
  }
  *(v4f*)(part + (size_t)(sblk * BATCH + b) * DIM + t * 4) = acc;
}

// ---------------------------------------------------------------------------
// Stage 2: reduce the SSPLIT partials -> m[b, d], scale by 1/S.
// ---------------------------------------------------------------------------
__global__ __launch_bounds__(256) void hcen_mean_stage2(
    const float* __restrict__ part, float* __restrict__ m) {
  const int i = blockIdx.x * 256 + threadIdx.x;   // 0..BATCH*DIM-1
  float acc = 0.f;
#pragma unroll 8
  for (int s = 0; s < SSPLIT; ++s)
    acc += part[(size_t)s * (BATCH * DIM) + i];
  m[i] = acc * (1.0f / (float)SEQ);
}

// ---------------------------------------------------------------------------
// GEMM: D[16,N] = A[16,K] @ W[N,K]^T + bias[N], one wave per 16x16 tile.
// blockDim must be 32 (wave32, EXEC all ones as WMMA requires).
// ---------------------------------------------------------------------------
__global__ __launch_bounds__(32) void hcen_gemm16(
    const float* __restrict__ A, const float* __restrict__ W,
    const float* __restrict__ bias, float* __restrict__ D,
    int K, int N) {
  const int n0   = blockIdx.x * 16;
  const int lane = threadIdx.x;         // 0..31
  const int half = lane >> 4;           // 0 or 1
  const int lm   = lane & 15;

  // C/D layout: vgpr j at lane -> D[j + 8*half, n0 + lm]; bias = f(col) only.
  v8f c;
  const float bv = bias[n0 + lm];
#pragma unroll
  for (int j = 0; j < 8; ++j) c[j] = bv;

#if __has_builtin(__builtin_amdgcn_wmma_f32_16x16x4_f32)
  // Full-precision fp32 WMMA path.
  // A 16x4 f32: lane holds A[lm, k0 + 2*half + {0,1}]  (contiguous pair)
  // B  4x16 f32: lane holds W[n0+lm, k0 + 2*half + {0,1}]
  const float* arow = A + (size_t)lm * K + 2 * half;
  const float* wrow = W + (size_t)(n0 + lm) * K + 2 * half;
#pragma unroll 8
  for (int k0 = 0; k0 < K; k0 += 4) {
    v2f av = *(const v2f*)(arow + k0);
    v2f bv2 = *(const v2f*)(wrow + k0);
    c = __builtin_amdgcn_wmma_f32_16x16x4_f32(
        false, av, false, bv2, (short)0, c, false, false);
  }
#else
  // Fallback: codegen-confirmed f16 WMMA (K-step 32), f32 accumulate.
  // 16-bit A 16x32 layout: element e -> k = k0 + 16*(e/8) + 8*half + (e%8)
  const float* arow = A + (size_t)lm * K + 8 * half;
  const float* wrow = W + (size_t)(n0 + lm) * K + 8 * half;
  for (int k0 = 0; k0 < K; k0 += 32) {
    v16h af, bf;
#pragma unroll
    for (int e = 0; e < 16; ++e) {
      const int off = k0 + 16 * (e >> 3) + (e & 7);
      af[e] = (_Float16)arow[off];
      bf[e] = (_Float16)wrow[off];
    }
    c = __builtin_amdgcn_wmma_f32_16x16x32_f16(
        false, af, false, bf, (short)0, c, false, false);
  }
#endif

#pragma unroll
  for (int j = 0; j < 8; ++j)
    D[(size_t)(j + 8 * half) * N + (n0 + lm)] = c[j];
}

// ---------------------------------------------------------------------------
// Launch: inputs (x, W_enc, b_enc, W_out, b_out) all fp32; out fp32 [16,1024].
// Workspace: partials (4MB) | m (64KB) | enc (64KB).
// ---------------------------------------------------------------------------
extern "C" void kernel_launch(void* const* d_in, const int* in_sizes, int n_in,
                              void* d_out, int out_size, void* d_ws, size_t ws_size,
                              hipStream_t stream) {
  const float* x     = (const float*)d_in[0];
  const float* W_enc = (const float*)d_in[1];
  const float* b_enc = (const float*)d_in[2];
  const float* W_out = (const float*)d_in[3];
  const float* b_out = (const float*)d_in[4];
  float* out = (float*)d_out;

  float* part = (float*)d_ws;                       // SSPLIT*B*DIM floats = 4MB
  float* m    = part + (size_t)SSPLIT * BATCH * DIM; // B*DIM floats
  float* enc  = m + (size_t)BATCH * DIM;             // B*DIM floats

  dim3 g1(SSPLIT, BATCH);
  hcen_mean_stage1<<<g1, 256, 0, stream>>>(x, part);
  hcen_mean_stage2<<<(BATCH * DIM) / 256, 256, 0, stream>>>(part, m);
  hcen_gemm16<<<DIM / 16, 32, 0, stream>>>(m,   W_enc, b_enc, enc, DIM, DIM);
  hcen_gemm16<<<DIM / 16, 32, 0, stream>>>(enc, W_out, b_out, out, DIM, DIM);
}
